// BitESNModule_30734785970270
// MI455X (gfx1250) — compile-verified, matrix operand order checked by reading, not run on hardware
//
#include <hip/hip_runtime.h>
#include <math.h>

constexpr int B   = 32;
constexpr int T   = 512;
constexpr int IN  = 256;
constexpr int OUT = 2048;
constexpr float ALPHA = 0.1f;

constexpr int NWG   = 32;
constexpr int SLICE = OUT / NWG;
constexpr int TPB   = 256;

constexpr size_t BAR_OFF = 0;
constexpr size_t ST_OFF  = 256;
constexpr size_t ST_BYTES = (size_t)B * OUT * 4;
constexpr size_t WQ_OFF  = ST_OFF + 2 * ST_BYTES;
constexpr size_t WIN_OFF = WQ_OFF + (size_t)OUT * OUT;

constexpr int LDS_WQ  = SLICE * OUT;
constexpr int LDS_Q   = B * OUT;
constexpr int LDS_XB  = B * IN * 2;
constexpr int LDS_TOTAL = LDS_WQ + LDS_Q + LDS_XB;

typedef __attribute__((ext_vector_type(4)))  int    i4;
typedef __attribute__((ext_vector_type(4)))  float  f4;
typedef __attribute__((ext_vector_type(8)))  int    v8i;
typedef __attribute__((ext_vector_type(8)))  float  v8f;
typedef __attribute__((ext_vector_type(16))) __bf16 v16bf;

union U16BF { i4 q[2]; v16bf v; };
union U8I   { i4 q[2]; v8i   v; };

__global__ void bitesn_pack_wres(const float* __restrict__ w, signed char* __restrict__ wq) {
    int idx = blockIdx.x * blockDim.x + threadIdx.x;
    int n = idx >> 11;
    int k = idx & (OUT - 1);
    wq[idx] = (signed char)(int)w[(size_t)k * OUT + n];
}

__global__ void bitesn_pack_win(const float* __restrict__ w, __bf16* __restrict__ wt) {
    int idx = blockIdx.x * blockDim.x + threadIdx.x;
    int n = idx >> 8;
    int k = idx & (IN - 1);
    wt[idx] = (__bf16)w[(size_t)k * OUT + n];
}

__global__ void bitesn_init(float* __restrict__ st, int* __restrict__ bar) {
    int i = blockIdx.x * blockDim.x + threadIdx.x;
    if (i < 2 * B * OUT) st[i] = 0.0f;
    if (i == 0) { bar[0] = 0; bar[1] = 0; }
}

__device__ __forceinline__ void grid_barrier(int* cnt, int* gen, int nwg) {
    __threadfence();
    __syncthreads();
    if (threadIdx.x == 0) {
        int g = __hip_atomic_load(gen, __ATOMIC_RELAXED, __HIP_MEMORY_SCOPE_AGENT);
        int prev = __hip_atomic_fetch_add(cnt, 1, __ATOMIC_ACQ_REL, __HIP_MEMORY_SCOPE_AGENT);
        if (prev == nwg - 1) {
            __hip_atomic_store(cnt, 0, __ATOMIC_RELAXED, __HIP_MEMORY_SCOPE_AGENT);
            __hip_atomic_fetch_add(gen, 1, __ATOMIC_RELEASE, __HIP_MEMORY_SCOPE_AGENT);
        } else {
            while (__hip_atomic_load(gen, __ATOMIC_ACQUIRE, __HIP_MEMORY_SCOPE_AGENT) == g) {
                __builtin_amdgcn_s_sleep(1);
            }
        }
    }
    __syncthreads();
    __threadfence();
}

__device__ __forceinline__ int clamp_i8(int v) {
    return min(127, max(-128, v));
}

__global__ void __launch_bounds__(TPB, 1)
bitesn_scan(const float* __restrict__ Xg,
            const signed char* __restrict__ wqg,
            const __bf16* __restrict__ winT,
            float* __restrict__ st0,
            float* __restrict__ st1,
            float* __restrict__ outg,
            int* __restrict__ bar) {
    extern __shared__ char smem[];
    signed char* wqld = (signed char*)smem;
    signed char* qld  = (signed char*)(smem + LDS_WQ);
    __bf16*      xb   = (__bf16*)(smem + LDS_WQ + LDS_Q);

    const int wg   = blockIdx.x;
    const int tid  = threadIdx.x;
    const int wave = tid >> 5;
    const int lane = tid & 31;
    const int mt = wave >> 2, nt = wave & 3;
    const int m0 = mt * 16,  n0 = nt * 16;
    const int lh = lane >> 4, ln = lane & 15;
    const int colg = wg * SLICE + n0 + ln;

    {
        const i4* src = (const i4*)(wqg + (size_t)wg * SLICE * OUT);
        i4* dst = (i4*)wqld;
        #pragma unroll 4
        for (int i = tid; i < LDS_WQ / 16; i += TPB) dst[i] = src[i];
    }
    __syncthreads();

    for (int t = 0; t < T; ++t) {
        const float* sin  = (t & 1) ? st1 : st0;
        float*       sout = (t & 1) ? st0 : st1;

        {
            const int r = tid >> 3, sub = tid & 7;
            const f4* srow = (const f4*)(sin + (size_t)r * OUT + sub * 256);
            float m = 0.0f;
            #pragma unroll 8
            for (int j = 0; j < 64; ++j) {
                f4 v = srow[j];
                m = fmaxf(m, fmaxf(fmaxf(fabsf(v.x), fabsf(v.y)),
                                   fmaxf(fabsf(v.z), fabsf(v.w))));
            }
            m = fmaxf(m, __shfl_xor(m, 1, 32));
            m = fmaxf(m, __shfl_xor(m, 2, 32));
            m = fmaxf(m, __shfl_xor(m, 4, 32));
            const float scale = 127.0f / fmaxf(m, 1e-5f);
            int* qrow = (int*)(qld + r * OUT + sub * 256);
            #pragma unroll 8
            for (int j = 0; j < 64; ++j) {
                f4 v = srow[j];
                int c0 = clamp_i8((int)rintf(v.x * scale));
                int c1 = clamp_i8((int)rintf(v.y * scale));
                int c2 = clamp_i8((int)rintf(v.z * scale));
                int c3 = clamp_i8((int)rintf(v.w * scale));
                qrow[j] = (c0 & 255) | ((c1 & 255) << 8) | ((c2 & 255) << 16) | ((c3 & 255) << 24);
            }
        }

        {
            const int e0 = tid * 32;
            const int b  = e0 >> 8, k0 = e0 & 255;
            const float* xp = Xg + ((size_t)b * T + t) * IN + k0;
            __bf16* xo = xb + b * IN + k0;
            #pragma unroll 8
            for (int j = 0; j < 32; j += 4) {
                f4 v = *(const f4*)(xp + j);
                xo[j]     = (__bf16)v.x;
                xo[j + 1] = (__bf16)v.y;
                xo[j + 2] = (__bf16)v.z;
                xo[j + 3] = (__bf16)v.w;
            }
            if (t + 1 < T) __builtin_prefetch(xp + IN, 0, 1);
        }
        __syncthreads();

        v8f u   = {0, 0, 0, 0, 0, 0, 0, 0};
        v8i acc = {0, 0, 0, 0, 0, 0, 0, 0};
        {
            const __bf16*      wcol = winT + (size_t)colg * IN;
            const signed char* wc   = wqld + (n0 + ln) * OUT;
            const signed char* qr   = qld  + (m0 + ln) * OUT;
            const __bf16*      xr   = xb   + (m0 + ln) * IN;

            #pragma unroll
            for (int kk = 0; kk < 8; ++kk) {
                {
                    const int k0 = kk * 32;
                    U16BF a, bm;
                    const i4* pa = (const i4*)(xr + k0 + lh * 16);
                    a.q[0] = pa[0]; a.q[1] = pa[1];
                    const i4* pb = (const i4*)(wcol + k0 + lh * 16);
                    bm.q[0] = pb[0]; bm.q[1] = pb[1];
                    u = __builtin_amdgcn_wmma_f32_16x16x32_bf16(
                            false, a.v, false, bm.v, (short)0, u, false, false);
                }
                #pragma unroll
                for (int j = 0; j < 4; ++j) {
                    const int k0 = kk * 256 + j * 64;
                    U8I a, b2;
                    const i4* pa = (const i4*)(qr + k0 + lh * 32);
                    a.q[0] = pa[0]; a.q[1] = pa[1];
                    const i4* pb = (const i4*)(wc + k0 + lh * 32);
                    b2.q[0] = pb[0]; b2.q[1] = pb[1];
                    acc = __builtin_amdgcn_wmma_i32_16x16x64_iu8(
                            true, a.v, true, b2.v, acc, false, false);
                }
            }
        }

        #pragma unroll
        for (int j = 0; j < 8; ++j) {
            const int row = m0 + lh * 8 + j;
            const float iv  = (float)acc[j] * (1.0f / 127.0f) + u[j];
            const float old = sin[(size_t)row * OUT + colg];
            const float nv  = (1.0f - ALPHA) * old + ALPHA * tanhf(iv);
            sout[(size_t)row * OUT + colg] = nv;
            outg[((size_t)row * T + t) * OUT + colg] = nv;
        }

        if (t + 1 < T) grid_barrier(&bar[0], &bar[1], NWG);
    }
}

extern "C" void kernel_launch(void* const* d_in, const int* in_sizes, int n_in,
                              void* d_out, int out_size, void* d_ws, size_t ws_size,
                              hipStream_t stream) {
    const float* X     = (const float*)d_in[0];
    const float* w_in  = (const float*)d_in[1];
    const float* w_res = (const float*)d_in[2];
    float* out = (float*)d_out;

    char* ws = (char*)d_ws;
    int*         bar  = (int*)(ws + BAR_OFF);
    float*       st   = (float*)(ws + ST_OFF);
    signed char* wq   = (signed char*)(ws + WQ_OFF);
    __bf16*      winT = (__bf16*)(ws + WIN_OFF);

    bitesn_init<<<(2 * B * OUT + TPB - 1) / TPB, TPB, 0, stream>>>(st, bar);
    bitesn_pack_wres<<<(OUT * OUT) / TPB, TPB, 0, stream>>>(w_res, wq);
    bitesn_pack_win<<<(OUT * IN) / TPB, TPB, 0, stream>>>(w_in, winT);

    bitesn_scan<<<NWG, TPB, LDS_TOTAL, stream>>>(
        X, wq, winT, st, st + (size_t)B * OUT, out, bar);
}
